// Recurrentv1MaskRCNNConvUpsampleHead_63900523430021
// MI455X (gfx1250) — matrix operand order repaired and checked by reading, not compiled, not run on hardware
//
#include <hip/hip_runtime.h>
#include <hip/hip_bf16.h>
#include <math.h>

typedef __bf16 bf16;
typedef __attribute__((ext_vector_type(16))) __bf16 bf16x16;
typedef __attribute__((ext_vector_type(8)))  __bf16 bf16x8;
typedef __attribute__((ext_vector_type(8)))  float  f32x8;

#define NROI 512
#define P14  (NROI*196)      // 100352 deconv-input pixels
#define MT14 (P14/16)        // 6272
#define Q28  (NROI*784)      // 401408 deconv-output pixels
#define P26  (NROI*676)      // 346112 state pixels
#define MB26 (P26/32)        // 10816 blocks of 32 pixels
#define MCH  80
#define CP   96

// ---------------- WMMA helpers ----------------

__device__ __forceinline__ f32x8 wmma_bf16(bf16x16 a, bf16x16 b, f32x8 c) {
  return __builtin_amdgcn_wmma_f32_16x16x32_bf16(false, a, false, b, (short)0, c, false, false);
}

// A-fragment (16x32 bf16) from LDS tile. Per ISA 7.12.2:
// lanes0-15: e0..7 -> K = 0..7,  e8..15 -> K = 16..23
// lanes16-31: e0..7 -> K = 8..15, e8..15 -> K = 24..31
__device__ __forceinline__ bf16x16 frag_from_lds(const bf16* sA, int m, int half,
                                                 int kbyte, int rowBytes) {
  const char* base = (const char*)sA + m * rowBytes + kbyte + half * 16;
  union { bf16x16 v; bf16x8 h[2]; } u;
  u.h[0] = *(const bf16x8*)(base);
  u.h[1] = *(const bf16x8*)(base + 32);
  return u.v;
}

// B-fragment (32x16 bf16): per-wave lane base + constant byte offset so the
// compiler folds `off` into the global_load_b128 immediate offset field.
__device__ __forceinline__ bf16x16 frag_w(const char* laneBase, int off) {
  union { bf16x16 v; bf16x8 h[2]; } u;
  u.h[0] = *(const bf16x8*)(laneBase + off);
  u.h[1] = *(const bf16x8*)(laneBase + off + 16);
  return u.v;
}

__device__ __forceinline__ float sigmoidf_(float v) { return 1.f / (1.f + __expf(-v)); }
__device__ __forceinline__ float softplusf_(float v) { return v > 20.f ? v : log1pf(__expf(v)); }

// ---------------- weight pre-swizzle ----------------
// dst layout: [tap][kb][nt][lane(32)][e(16)] bf16.
// B matrix lane mapping: n = nt*16 + (lane&15); k = kb*32 + (lane>>4)*8 + (e&7) + (e>>3)*16.
__global__ __launch_bounds__(256) void prep_weights(const float* __restrict__ src,
                                                    bf16* __restrict__ dst,
                                                    int O, int I, int taps, int KB, int NT,
                                                    int kmajor) {
  int idx = blockIdx.x * 256 + threadIdx.x;
  int total = taps * KB * NT * 512;
  if (idx >= total) return;
  int e    = idx & 15;
  int lane = (idx >> 4) & 31;
  int blk  = idx >> 9;
  int nt  = blk % NT;
  int tmp = blk / NT;
  int kb  = tmp % KB;
  int tap = tmp / KB;
  int k = kb * 32 + (lane >> 4) * 8 + (e & 7) + (e >> 3) * 16;
  int n = nt * 16 + (lane & 15);
  float v = 0.f;
  if (n < O && k < I)
    v = kmajor ? src[((size_t)k * O + n) * taps + tap]
               : src[((size_t)n * I + k) * taps + tap];
  dst[idx] = (bf16)v;
}

// ---------------- NCHW f32 -> pixel-major bf16 ----------------
__global__ __launch_bounds__(256) void transpose_x(const float* __restrict__ x,
                                                   bf16* __restrict__ xt) {
  int idx = blockIdx.x * 256 + threadIdx.x;     // over 512*256*196
  int rr = idx % 196;
  int t  = idx / 196;
  int c  = t % 256;
  int nimg = t / 256;
  xt[((size_t)nimg * 196 + rr) * 256 + c] = (bf16)x[idx];
}

// ---------------- deconv 2x2 stride-2 as 4 GEMMs ----------------
// xt: [P14][256] bf16 ; out xd: [Q28][256] bf16 (pixel-major 28x28)
__global__ __launch_bounds__(128) void deconv_gemm(const bf16* __restrict__ xt,
                                                   const bf16* __restrict__ W,
                                                   const float* __restrict__ bias,
                                                   bf16* __restrict__ xd) {
  __shared__ uint4 sA4[16 * 33];                // 16 rows x 32 chunks (+1 pad)
  const bf16* sA = (const bf16*)sA4;
  int tid = threadIdx.x;
  int wid = tid >> 5, lane = tid & 31;
  int half = lane >> 4, nl = lane & 15;
  int pbase = blockIdx.x * 16;
  int ntile = blockIdx.y * 4 + wid;
  const char* wlane = (const char*)W + ntile * 1024 + lane * 32;

  for (int idx = tid; idx < 512; idx += 128) {
    int row = idx >> 5, ch = idx & 31;
    sA4[row * 33 + ch] = ((const uint4*)(xt + (size_t)(pbase + row) * 256))[ch];
  }
  __syncthreads();

  f32x8 acc[4] = {};
#pragma unroll
  for (int kb = 0; kb < 8; ++kb) {
    bf16x16 a = frag_from_lds(sA, nl, half, kb * 64, 33 * 16);
#pragma unroll
    for (int tap = 0; tap < 4; ++tap) {
      bf16x16 b = frag_w(wlane, (tap * 8 + kb) * 16384);
      acc[tap] = wmma_bf16(a, b, acc[tap]);
    }
  }

  int cbase = ntile * 16;
  float bv = bias[cbase + nl];
#pragma unroll
  for (int tap = 0; tap < 4; ++tap) {
    int ta = tap >> 1, tb = tap & 1;
#pragma unroll
    for (int r = 0; r < 8; ++r) {
      int p = pbase + r + half * 8;
      int nimg = p / 196, rr = p % 196;
      int i = rr / 14, j = rr % 14;
      int q = nimg * 784 + (2 * i + ta) * 28 + (2 * j + tb);
      xd[(size_t)q * 256 + cbase + nl] = (bf16)(acc[tap][r] + bv);
    }
  }
}

// ---------------- preproc 3x3 VALID 256->80 ----------------
// xd: [Q28][256] bf16 ; out xpre: [P26][80] f32.  M-tile = 32 pixels.
__global__ __launch_bounds__(192) void preproc_gemm(const bf16* __restrict__ xd,
                                                    const bf16* __restrict__ W,
                                                    const float* __restrict__ bias,
                                                    float* __restrict__ xpre) {
  __shared__ uint4 sA4[32 * 33];
  const bf16* sA = (const bf16*)sA4;
  int tid = threadIdx.x;
  int wid = tid >> 5, lane = tid & 31;
  int half = lane >> 4, nl = lane & 15;
  int pbase = blockIdx.x * 32;
  const char* wlane = (const char*)W + wid * 1024 + lane * 32;

  f32x8 acc[2] = {};
  for (int tap = 0; tap < 9; ++tap) {
    int dh = tap / 3, dw = tap % 3;
    const char* wt = wlane + tap * 49152;
    __builtin_prefetch(wt + 49152, 0, 0);
    for (int idx = tid; idx < 1024; idx += 192) {
      int row = idx >> 5, ch = idx & 31;
      int p = pbase + row;
      int nimg = p / 676, rr = p % 676;
      int i = rr / 26, j = rr % 26;
      int q = nimg * 784 + (i + dh) * 28 + (j + dw);
      sA4[row * 33 + ch] = ((const uint4*)(xd + (size_t)q * 256))[ch];
    }
    __syncthreads();
#pragma unroll
    for (int kb = 0; kb < 8; ++kb) {
      bf16x16 b = frag_w(wt, kb * 6144);
      bf16x16 a0 = frag_from_lds(sA, nl, half, kb * 64, 33 * 16);
      bf16x16 a1 = frag_from_lds(sA, 16 + nl, half, kb * 64, 33 * 16);
      acc[0] = wmma_bf16(a0, b, acc[0]);
      acc[1] = wmma_bf16(a1, b, acc[1]);
    }
    __syncthreads();
  }

  int c = wid * 16 + nl;
  if (c < MCH) {
    float bv = bias[c];
#pragma unroll
    for (int ms = 0; ms < 2; ++ms) {
#pragma unroll
      for (int r = 0; r < 8; ++r) {
        int p = pbase + ms * 16 + r + half * 8;
        xpre[(size_t)p * MCH + c] = acc[ms][r] + bv;
      }
    }
  }
}

// ---------------- step 1: g1 = sigmoid(h*U1 + b1); hg = h * g1 ----------------
__global__ __launch_bounds__(192) void step_gate1(const bf16* __restrict__ h,
                                                  const bf16* __restrict__ Wu1,
                                                  const float* __restrict__ u1b,
                                                  bf16* __restrict__ hg) {
  __shared__ uint4 sA4[32 * 13];               // 32 rows x 12 chunks (+1 pad)
  const bf16* sA = (const bf16*)sA4;
  int tid = threadIdx.x;
  int wid = tid >> 5, lane = tid & 31;
  int half = lane >> 4, nl = lane & 15;
  int pbase = blockIdx.x * 32;
  const char* wlane = (const char*)Wu1 + wid * 1024 + lane * 32;

  for (int idx = tid; idx < 384; idx += 192) {
    int row = idx / 12, ch = idx % 12;
    sA4[row * 13 + ch] = ((const uint4*)(h + (size_t)(pbase + row) * CP))[ch];
  }
  __syncthreads();

  f32x8 acc[2] = {};
#pragma unroll
  for (int kb = 0; kb < 3; ++kb) {
    bf16x16 b = frag_w(wlane, kb * 6144);
    bf16x16 a0 = frag_from_lds(sA, nl, half, kb * 64, 13 * 16);
    bf16x16 a1 = frag_from_lds(sA, 16 + nl, half, kb * 64, 13 * 16);
    acc[0] = wmma_bf16(a0, b, acc[0]);
    acc[1] = wmma_bf16(a1, b, acc[1]);
  }

  int c = wid * 16 + nl;
  float bv = (c < MCH) ? u1b[c] : 0.f;
#pragma unroll
  for (int ms = 0; ms < 2; ++ms) {
#pragma unroll
    for (int r = 0; r < 8; ++r) {
      int p = pbase + ms * 16 + r + half * 8;
      float g = sigmoidf_(acc[ms][r] + bv);
      float hv = (float)h[(size_t)p * CP + c];
      hg[(size_t)p * CP + c] = (bf16)(hv * g);
    }
  }
}

// ---------------- step 2: c1 = conv3x3(hg); ns1 = softplus(x - c1*(a*h+mu)) ----------------
__global__ __launch_bounds__(192) void step_inh(const bf16* __restrict__ hg,
                                                const bf16* __restrict__ Winh,
                                                const bf16* __restrict__ h,
                                                const float* __restrict__ xpre,
                                                const float* __restrict__ alpha,
                                                const float* __restrict__ mu,
                                                bf16* __restrict__ ns1) {
  __shared__ uint4 sA4[32 * 13];
  const bf16* sA = (const bf16*)sA4;
  int tid = threadIdx.x;
  int wid = tid >> 5, lane = tid & 31;
  int half = lane >> 4, nl = lane & 15;
  int pbase = blockIdx.x * 32;
  const char* wlane = (const char*)Winh + wid * 1024 + lane * 32;

  f32x8 acc[2] = {};
  for (int tap = 0; tap < 9; ++tap) {
    int dh = tap / 3 - 1, dw = tap % 3 - 1;
    const char* wt = wlane + tap * 18432;
    __builtin_prefetch(wt + 18432, 0, 0);
    for (int idx = tid; idx < 384; idx += 192) {
      int row = idx / 12, ch = idx % 12;
      int p = pbase + row;
      int nimg = p / 676, rr = p % 676;
      int i = rr / 26, j = rr % 26;
      int ii = i + dh, jj = j + dw;
      uint4 v = {0u, 0u, 0u, 0u};
      if ((unsigned)ii < 26u && (unsigned)jj < 26u)
        v = ((const uint4*)(hg + (size_t)(nimg * 676 + ii * 26 + jj) * CP))[ch];
      sA4[row * 13 + ch] = v;
    }
    __syncthreads();
#pragma unroll
    for (int kb = 0; kb < 3; ++kb) {
      bf16x16 b = frag_w(wt, kb * 6144);
      bf16x16 a0 = frag_from_lds(sA, nl, half, kb * 64, 13 * 16);
      bf16x16 a1 = frag_from_lds(sA, 16 + nl, half, kb * 64, 13 * 16);
      acc[0] = wmma_bf16(a0, b, acc[0]);
      acc[1] = wmma_bf16(a1, b, acc[1]);
    }
    __syncthreads();
  }

  int c = wid * 16 + nl;
  float al = (c < MCH) ? alpha[c] : 0.f;
  float m_ = (c < MCH) ? mu[c] : 0.f;
#pragma unroll
  for (int ms = 0; ms < 2; ++ms) {
#pragma unroll
    for (int r = 0; r < 8; ++r) {
      int p = pbase + ms * 16 + r + half * 8;
      float hv = (float)h[(size_t)p * CP + c];
      float xv = (c < MCH) ? xpre[(size_t)p * MCH + c] : 0.f;
      float ns = softplusf_(xv - acc[ms][r] * (al * hv + m_));
      if (c >= MCH) ns = 0.f;
      ns1[(size_t)p * CP + c] = (bf16)ns;
    }
  }
}

// ---------------- step 3: c2 = conv3x3(ns1); g2 = sigmoid(ns1*U2+b2);
//                  h = (1-g2)*h + g2*softplus(c2)  (fused 3x3 + 1x1 GEMM) ----------------
__global__ __launch_bounds__(192) void step_exc(const bf16* __restrict__ ns1,
                                                const bf16* __restrict__ Wexc,
                                                const bf16* __restrict__ Wu2,
                                                const float* __restrict__ u2b,
                                                bf16* __restrict__ h,
                                                float* __restrict__ out,
                                                int last) {
  __shared__ uint4 sA4[32 * 13];
  const bf16* sA = (const bf16*)sA4;
  int tid = threadIdx.x;
  int wid = tid >> 5, lane = tid & 31;
  int half = lane >> 4, nl = lane & 15;
  int pbase = blockIdx.x * 32;
  const char* wlane  = (const char*)Wexc + wid * 1024 + lane * 32;
  const char* wu2ln  = (const char*)Wu2  + wid * 1024 + lane * 32;

  f32x8 accc[2] = {}, accg[2] = {};
  for (int tap = 0; tap < 9; ++tap) {
    int dh = tap / 3 - 1, dw = tap % 3 - 1;
    const char* wt = wlane + tap * 18432;
    __builtin_prefetch(wt + 18432, 0, 0);
    for (int idx = tid; idx < 384; idx += 192) {
      int row = idx / 12, ch = idx % 12;
      int p = pbase + row;
      int nimg = p / 676, rr = p % 676;
      int i = rr / 26, j = rr % 26;
      int ii = i + dh, jj = j + dw;
      uint4 v = {0u, 0u, 0u, 0u};
      if ((unsigned)ii < 26u && (unsigned)jj < 26u)
        v = ((const uint4*)(ns1 + (size_t)(nimg * 676 + ii * 26 + jj) * CP))[ch];
      sA4[row * 13 + ch] = v;
    }
    __syncthreads();
#pragma unroll
    for (int kb = 0; kb < 3; ++kb) {
      bf16x16 b = frag_w(wt, kb * 6144);
      bf16x16 a0 = frag_from_lds(sA, nl, half, kb * 64, 13 * 16);
      bf16x16 a1 = frag_from_lds(sA, 16 + nl, half, kb * 64, 13 * 16);
      accc[0] = wmma_bf16(a0, b, accc[0]);
      accc[1] = wmma_bf16(a1, b, accc[1]);
      if (tap == 4) {                           // center tap == the 1x1 conv input
        bf16x16 b2 = frag_w(wu2ln, kb * 6144);
        accg[0] = wmma_bf16(a0, b2, accg[0]);
        accg[1] = wmma_bf16(a1, b2, accg[1]);
      }
    }
    __syncthreads();
  }

  int c = wid * 16 + nl;
  float bv = (c < MCH) ? u2b[c] : 0.f;
#pragma unroll
  for (int ms = 0; ms < 2; ++ms) {
#pragma unroll
    for (int r = 0; r < 8; ++r) {
      int p = pbase + ms * 16 + r + half * 8;
      float g2 = sigmoidf_(accg[ms][r] + bv);
      float h2 = softplusf_(accc[ms][r]);
      float hv = (float)h[(size_t)p * CP + c];
      float hn = (1.f - g2) * hv + g2 * h2;
      if (c >= MCH) hn = 0.f;
      h[(size_t)p * CP + c] = (bf16)hn;
      if (last && c < MCH) {
        int nimg = p / 676, rr = p % 676;
        out[((size_t)nimg * MCH + c) * 676 + rr] = hn;   // NCHW f32 output
      }
    }
  }
}

// ---------------- host launch ----------------
extern "C" void kernel_launch(void* const* d_in, const int* in_sizes, int n_in,
                              void* d_out, int out_size, void* d_ws, size_t ws_size,
                              hipStream_t stream) {
  (void)in_sizes; (void)n_in; (void)out_size; (void)ws_size;
  const float* x         = (const float*)d_in[0];
  const float* deconv_w  = (const float*)d_in[1];
  const float* deconv_b  = (const float*)d_in[2];
  const float* preproc_w = (const float*)d_in[3];
  const float* preproc_b = (const float*)d_in[4];
  const float* u1_w      = (const float*)d_in[5];
  const float* u1_b      = (const float*)d_in[6];
  const float* u2_w      = (const float*)d_in[7];
  const float* u2_b      = (const float*)d_in[8];
  const float* w_inh     = (const float*)d_in[9];
  const float* w_exc     = (const float*)d_in[10];
  const float* alpha     = (const float*)d_in[11];
  const float* mu        = (const float*)d_in[12];
  float* out = (float*)d_out;

  char* ws = (char*)d_ws;
  size_t off = 0;
  auto take = [&](size_t bytes) -> char* {
    char* p = ws + off;
    off += (bytes + 255) & ~(size_t)255;
    return p;
  };
  bf16* wDec  = (bf16*)take((size_t)4 * 8 * 16 * 512 * 2);
  bf16* wPre  = (bf16*)take((size_t)9 * 8 * 6 * 512 * 2);
  bf16* wU1   = (bf16*)take((size_t)3 * 6 * 512 * 2);
  bf16* wU2   = (bf16*)take((size_t)3 * 6 * 512 * 2);
  bf16* wInh  = (bf16*)take((size_t)9 * 3 * 6 * 512 * 2);
  bf16* wExc  = (bf16*)take((size_t)9 * 3 * 6 * 512 * 2);
  bf16* xt    = (bf16*)take((size_t)P14 * 256 * 2);
  bf16* xd    = (bf16*)take((size_t)Q28 * 256 * 2);
  float* xpre = (float*)take((size_t)P26 * MCH * 4);
  bf16* hbuf  = (bf16*)take((size_t)P26 * CP * 2);
  bf16* hg    = (bf16*)take((size_t)P26 * CP * 2);
  bf16* nsb   = (bf16*)take((size_t)P26 * CP * 2);

  auto blocks = [](int total) { return dim3((unsigned)((total + 255) / 256)); };
  prep_weights<<<blocks(4 * 8 * 16 * 512), 256, 0, stream>>>(deconv_w, wDec, 256, 256, 4, 8, 16, 1);
  prep_weights<<<blocks(9 * 8 * 6 * 512), 256, 0, stream>>>(preproc_w, wPre, 80, 256, 9, 8, 6, 0);
  prep_weights<<<blocks(3 * 6 * 512), 256, 0, stream>>>(u1_w, wU1, 80, 80, 1, 3, 6, 0);
  prep_weights<<<blocks(3 * 6 * 512), 256, 0, stream>>>(u2_w, wU2, 80, 80, 1, 3, 6, 0);
  prep_weights<<<blocks(9 * 3 * 6 * 512), 256, 0, stream>>>(w_inh, wInh, 80, 80, 9, 3, 6, 0);
  prep_weights<<<blocks(9 * 3 * 6 * 512), 256, 0, stream>>>(w_exc, wExc, 80, 80, 9, 3, 6, 0);

  transpose_x<<<dim3(NROI * 256 * 196 / 256), 256, 0, stream>>>(x, xt);
  deconv_gemm<<<dim3(MT14, 4), 128, 0, stream>>>(xt, wDec, deconv_b, xd);
  preproc_gemm<<<dim3(MB26), 192, 0, stream>>>(xd, wPre, preproc_b, xpre);

  hipMemsetAsync(hbuf, 0, (size_t)P26 * CP * 2, stream);

  for (int t = 0; t < 10; ++t) {
    step_gate1<<<dim3(MB26), 192, 0, stream>>>(hbuf, wU1, u1_b, hg);
    step_inh<<<dim3(MB26), 192, 0, stream>>>(hg, wInh, hbuf, xpre, alpha, mu, nsb);
    step_exc<<<dim3(MB26), 192, 0, stream>>>(nsb, wExc, wU2, u2_b, hbuf, out, t == 9);
  }
}
